// GATConv_31817117729572
// MI455X (gfx1250) — compile-verified, matrix operand order checked by reading, not run on hardware
//
#include <hip/hip_runtime.h>

// ---------------------------------------------------------------------------
// GATConv on MI455X (gfx1250, wave32).
//   1) ft = feat @ W via v_wmma_f32_16x16x32_bf16. W is pre-packed ONCE into
//      bf16 lane-fragment layout (128 KB); the 16x256 A row-panel is staged
//      into LDS once per block (float4 global loads -> packed bf16 -> b128 ds
//      stores), then the K loop is fully unrolled: 8 WMMAs per wave, fed by
//      ds_load_b128 (A) + global_load_b128 (packed W).
//   2) el/er node logits: wave-per-node dots + shfl reductions.
//   3) segmented softmax over dst via L2-resident atomics; edge logits are
//      recomputed from el/er (1.6 MB, L2) instead of materializing [E,H].
//   4) scatter: wave-per-edge float4 gathers, atomicAdd f32 into rst
//      (102.4 MB -> fits in 192 MB L2, atomics stay on-chip).
// ---------------------------------------------------------------------------

typedef __attribute__((ext_vector_type(16))) __bf16 v16bf;
typedef __attribute__((ext_vector_type(8)))  float  v8f;

#define IN_FEATS 256
#define HD 256            // NUM_HEADS * OUT_FEATS
#define KSTEPS 8          // 256 / 32
#define NEG_SLOPE 0.2f

union FragBF { v16bf v; uint4 q[2]; };

__device__ __forceinline__ unsigned short f2bf_u16(float f) {
  union { float f; unsigned u; } in; in.f = f;
  unsigned u = in.u;
  return (unsigned short)((u + 0x7FFFu + ((u >> 16) & 1u)) >> 16);  // RNE
}
__device__ __forceinline__ unsigned pack2bf(float a, float b) {
  return (unsigned)f2bf_u16(a) | ((unsigned)f2bf_u16(b) << 16);
}

__device__ __forceinline__ unsigned enc_ord(float v) {
  unsigned b = __float_as_uint(v);
  return (b & 0x80000000u) ? ~b : (b | 0x80000000u);
}
__device__ __forceinline__ float dec_ord(unsigned k) {
  unsigned b = (k & 0x80000000u) ? (k & 0x7FFFFFFFu) : ~k;
  return __uint_as_float(b);
}

// --------------------------- zero init -------------------------------------
__global__ __launch_bounds__(256) void zero_f32(float* __restrict__ p, size_t n) {
  size_t i = (size_t)blockIdx.x * 256 + threadIdx.x;
  if (i < n) p[i] = 0.0f;
}

// ------------------ one-time W pre-pack into WMMA fragments ----------------
// Layout: Wp[((nt*8 + ks)*32 + lane)*8 + u]  (uint = 2 bf16), nt = n-tile
// (16 tiles of 16 cols), ks = K step (32), lane fragment = 16 bf16:
// elems 0..7 -> K = khalf+0..7, elems 8..15 -> K = khalf+16..23,
// khalf = (lane>>4)*8, col = nt*16 + (lane&15).  Total 32768 uints = 128 KB.
__global__ __launch_bounds__(256) void pack_W(const float* __restrict__ W,
                                              unsigned* __restrict__ Wp) {
  int t = blockIdx.x * 256 + threadIdx.x;
  if (t >= 16 * KSTEPS * 32 * 8) return;
  int u    = t & 7;
  int lane = (t >> 3) & 31;
  int ks   = (t >> 8) & 7;
  int nt   = t >> 11;
  int col   = nt * 16 + (lane & 15);
  int khalf = (lane >> 4) * 8;
  int j0 = u * 2, j1 = u * 2 + 1;
  int k0 = ks * 32 + khalf + (j0 < 8 ? j0 : j0 + 8);
  int k1 = ks * 32 + khalf + (j1 < 8 ? j1 : j1 + 8);
  Wp[t] = pack2bf(W[(size_t)k0 * HD + col], W[(size_t)k1 * HD + col]);
}

// --------------------------- GEMM: ft = feat @ W ---------------------------
// Block = 256 threads = 8 waves. Block covers rows [m0,m0+16) x 128 cols;
// each wave owns one 16x16 output tile. Whole 16x256 A panel staged in LDS
// as bf16 once, then 8 fully-unrolled WMMA K-steps.
__global__ __launch_bounds__(256) void gat_gemm_wmma(
    const float* __restrict__ feat, const unsigned* __restrict__ Wp,
    float* __restrict__ ft, int N) {
  __shared__ unsigned AsU[16 * 128];        // 16 rows x 256 bf16 = 8 KB

  const int wave = threadIdx.x >> 5;
  const int lane = threadIdx.x & 31;
  const int m0   = (blockIdx.x >> 1) * 16;
  const int nt   = (blockIdx.x & 1) * 8 + wave;   // n-tile index, col0 = nt*16

  // ---- Stage A panel: thread t handles row (t>>4), 16 cols ((t&15)*16). ----
  {
    int m    = threadIdx.x >> 4;
    int kblk = (threadIdx.x & 15) * 16;
    int row  = m0 + m; if (row >= N) row = N - 1;
    const float4* src4 = (const float4*)(feat + (size_t)row * IN_FEATS + kblk);
    float4 f0 = src4[0], f1 = src4[1], f2 = src4[2], f3 = src4[3];
    uint4 p0, p1;
    p0.x = pack2bf(f0.x, f0.y); p0.y = pack2bf(f0.z, f0.w);
    p0.z = pack2bf(f1.x, f1.y); p0.w = pack2bf(f1.z, f1.w);
    p1.x = pack2bf(f2.x, f2.y); p1.y = pack2bf(f2.z, f2.w);
    p1.z = pack2bf(f3.x, f3.y); p1.w = pack2bf(f3.z, f3.w);
    uint4* dst4 = (uint4*)(AsU + m * 128 + (kblk >> 1));
    dst4[0] = p0; dst4[1] = p1;
  }
  // Prefetch this block's packed-W panel while waiting at the barrier.
  __builtin_prefetch(Wp + ((size_t)nt * KSTEPS * 32 + lane) * 8, 0, 0);
  __syncthreads();

  const int mrow  = lane & 15;
  const int khalf = (lane >> 4) * 8;        // 0 or 8
  const unsigned* abase = AsU + mrow * 128 + (khalf >> 1);
  const uint4*    wbase = (const uint4*)(Wp + ((size_t)nt * KSTEPS + 0) * 32 * 8) + lane * 2;

  v8f c = {};
#pragma unroll
  for (int ks = 0; ks < KSTEPS; ++ks) {
    FragBF a, b;
    const uint4* ap = (const uint4*)(abase + ks * 16);
    a.q[0] = ap[0];                          // K = khalf+0..7
    a.q[1] = ap[2];                          // K = khalf+16..23 (+8 uints)
    const uint4* wp = wbase + (size_t)ks * 64;  // 32 lanes * 2 uint4 per step
    b.q[0] = wp[0];
    b.q[1] = wp[1];
    c = __builtin_amdgcn_wmma_f32_16x16x32_bf16(
        /*neg_a=*/false, a.v, /*neg_b=*/false, b.v,
        /*c_mod=*/(short)0, c, /*reuse_a=*/false, /*reuse_b=*/false);
  }

  // C/D layout: VGPR v, lanes 0-15 -> M=v, lanes 16-31 -> M=v+8; N = lane&15.
  const int col = nt * 16 + (lane & 15);
#pragma unroll
  for (int v = 0; v < 8; ++v) {
    int row = m0 + ((lane < 16) ? v : v + 8);
    if (row < N) ft[(size_t)row * HD + col] = c[v];
  }
}

// ------------------- per-node attention logits el/er -----------------------
__global__ __launch_bounds__(256) void attn_logits(
    const float* __restrict__ ft, const float* __restrict__ al,
    const float* __restrict__ ar, float* __restrict__ el,
    float* __restrict__ er, int N) {
  const int wave = threadIdx.x >> 5;
  const int lane = threadIdx.x & 31;
  const int n = blockIdx.x * 8 + wave;
  if (n >= N) return;
  const int h = lane >> 3;
  const int off = (lane & 7) * 8;

  const float4* fp = (const float4*)(ft + (size_t)n * HD + lane * 8);
  float4 x0 = fp[0], x1 = fp[1];
  const float4* lp = (const float4*)(al + h * 64 + off);
  float4 l0 = lp[0], l1 = lp[1];
  const float4* rp = (const float4*)(ar + h * 64 + off);
  float4 r0 = rp[0], r1 = rp[1];

  float sl = x0.x*l0.x + x0.y*l0.y + x0.z*l0.z + x0.w*l0.w
           + x1.x*l1.x + x1.y*l1.y + x1.z*l1.z + x1.w*l1.w;
  float sr = x0.x*r0.x + x0.y*r0.y + x0.z*r0.z + x0.w*r0.w
           + x1.x*r1.x + x1.y*r1.y + x1.z*r1.z + x1.w*r1.w;
#pragma unroll
  for (int m = 1; m < 8; m <<= 1) {
    sl += __shfl_xor(sl, m, 8);
    sr += __shfl_xor(sr, m, 8);
  }
  if ((lane & 7) == 0) {
    el[n * 4 + h] = sl;
    er[n * 4 + h] = sr;
  }
}

// ------------------------- edge passes -------------------------------------
__device__ __forceinline__ float edge_logit(const float* el, const float* er,
                                            int s, int d, int h) {
  float v = el[s * 4 + h] + er[d * 4 + h];
  return v > 0.0f ? v : NEG_SLOPE * v;
}

__global__ __launch_bounds__(256) void edge_max(
    const int* __restrict__ src, const int* __restrict__ dst,
    const float* __restrict__ el, const float* __restrict__ er,
    unsigned* __restrict__ smax, int E) {
  int e = blockIdx.x * 256 + threadIdx.x;
  if (e >= E) return;
  int s = src[e], d = dst[e];
#pragma unroll
  for (int h = 0; h < 4; ++h)
    atomicMax(&smax[d * 4 + h], enc_ord(edge_logit(el, er, s, d, h)));
}

__global__ __launch_bounds__(256) void edge_expsum(
    const int* __restrict__ src, const int* __restrict__ dst,
    const float* __restrict__ el, const float* __restrict__ er,
    const unsigned* __restrict__ smax, float* __restrict__ ssum, int E) {
  int e = blockIdx.x * 256 + threadIdx.x;
  if (e >= E) return;
  int s = src[e], d = dst[e];
#pragma unroll
  for (int h = 0; h < 4; ++h) {
    float v  = edge_logit(el, er, s, d, h);
    float ex = __expf(v - dec_ord(smax[d * 4 + h]));
    atomicAdd(&ssum[d * 4 + h], ex);
  }
}

// Wave-per-edge weighted scatter: lane owns 8 contiguous features of one head.
__global__ __launch_bounds__(256) void edge_scatter(
    const int* __restrict__ src, const int* __restrict__ dst,
    const float* __restrict__ el, const float* __restrict__ er,
    const unsigned* __restrict__ smax, const float* __restrict__ ssum,
    const float* __restrict__ ft, float* __restrict__ rst, int E) {
  const int wave = threadIdx.x >> 5;
  const int lane = threadIdx.x & 31;
  const int e = blockIdx.x * 8 + wave;
  if (e >= E) return;
  const int s = src[e], d = dst[e];
  const int h = lane >> 3;

  float v = edge_logit(el, er, s, d, h);
  float a = __expf(v - dec_ord(smax[d * 4 + h])) / ssum[d * 4 + h];

  const float4* fp = (const float4*)(ft + (size_t)s * HD + lane * 8);
  float4 x0 = fp[0], x1 = fp[1];
  float* op = rst + (size_t)d * HD + lane * 8;
  atomicAdd(op + 0, x0.x * a); atomicAdd(op + 1, x0.y * a);
  atomicAdd(op + 2, x0.z * a); atomicAdd(op + 3, x0.w * a);
  atomicAdd(op + 4, x1.x * a); atomicAdd(op + 5, x1.y * a);
  atomicAdd(op + 6, x1.z * a); atomicAdd(op + 7, x1.w * a);
}

// ---------------------------------------------------------------------------
extern "C" void kernel_launch(void* const* d_in, const int* in_sizes, int n_in,
                              void* d_out, int out_size, void* d_ws, size_t ws_size,
                              hipStream_t stream) {
  const float* feat = (const float*)d_in[0];   // [N, 256]
  const float* W    = (const float*)d_in[1];   // [256, 256]
  const float* al   = (const float*)d_in[2];   // [1, 4, 64]
  const float* ar   = (const float*)d_in[3];   // [1, 4, 64]
  const int*   src  = (const int*)d_in[4];     // [E]
  const int*   dst  = (const int*)d_in[5];     // [E]

  const int N = in_sizes[0] / IN_FEATS;
  const int E = in_sizes[4];

  // Workspace: ft [N*256] | el [N*4] | er [N*4] | smax [N*4] u32 | ssum [N*4]
  //            | Wp [32768] u32 (packed bf16 W fragments)
  float*    ft   = (float*)d_ws;
  float*    el   = ft + (size_t)N * HD;
  float*    er   = el + (size_t)N * 4;
  unsigned* smax = (unsigned*)(er + (size_t)N * 4);
  float*    ssum = (float*)(smax + (size_t)N * 4);
  unsigned* Wp   = (unsigned*)(ssum + (size_t)N * 4);

  float* rst = (float*)d_out;
  const size_t outN = (size_t)out_size;
  const size_t nh   = (size_t)N * 4;

  // One-time W fragment pack (32768 uints).
  pack_W<<<128, 256, 0, stream>>>(W, Wp);

  // Init: rst = 0, ssum = 0, smax = key 0 (== encoded below any finite value).
  zero_f32<<<(unsigned)((outN + 255) / 256), 256, 0, stream>>>(rst, outN);
  zero_f32<<<(unsigned)((nh + 255) / 256), 256, 0, stream>>>(ssum, nh);
  zero_f32<<<(unsigned)((nh + 255) / 256), 256, 0, stream>>>((float*)smax, nh);

  // 1) Projection via WMMA.
  const int mTiles = (N + 15) / 16;
  gat_gemm_wmma<<<mTiles * 2, 256, 0, stream>>>(feat, Wp, ft, N);

  // 2) Node attention logits.
  attn_logits<<<(N + 7) / 8, 256, 0, stream>>>(ft, al, ar, el, er, N);

  // 3) Segmented softmax (max, then exp-sum).
  edge_max<<<(E + 255) / 256, 256, 0, stream>>>(src, dst, el, er, smax, E);
  edge_expsum<<<(E + 255) / 256, 256, 0, stream>>>(src, dst, el, er, smax, ssum, E);

  // 4) Weighted scatter aggregation.
  edge_scatter<<<(E + 7) / 8, 256, 0, stream>>>(src, dst, el, er, smax, ssum, ft, rst, E);
}
